// MyViT_40432822124651
// MI455X (gfx1250) — compile-verified, matrix operand order checked by reading, not run on hardware
//
#include <hip/hip_runtime.h>
#include <hip/hip_bf16.h>
#include <cstddef>

// ---------------- types ----------------
typedef __attribute__((ext_vector_type(16))) __bf16        v16bf;
typedef __attribute__((ext_vector_type(8)))  float         v8f;
typedef __attribute__((ext_vector_type(8)))  unsigned int  v8u;

union FragBF { v16bf v; v8u u; };
union V16B   { uint4 q; unsigned short s[8]; };

__device__ __forceinline__ unsigned short f2bf(float f) {
  unsigned int u = __float_as_uint(f);
  u += 0x7FFFu + ((u >> 16) & 1u);          // round-to-nearest-even
  return (unsigned short)(u >> 16);
}
__device__ __forceinline__ int imin(int a, int b) { return a < b ? a : b; }

// ---------------- generic WMMA GEMM ----------------
// C[m,n] = sum_k A[m,k]*B[n,k]           (bTrans=0, torch-Linear weight layout)
//        = sum_k A[m,k]*B[k,n]           (bTrans=1, for P @ V)
// + bias[n] + Cres[m,n], optional exact GELU; output fp32 (Cf) and/or bf16 (Cb).
// Batch z decomposes as zb=z/zInner, zi=z%zInner with independent strides.
struct GemmP {
  const unsigned short* A; long long aB0, aB1; int lda;
  const unsigned short* B; long long bB0, bB1; int ldb; int bTrans;
  const float* bias;       long long biasB0, biasB1;
  const float* Cres;       long long crB0, crB1; int ldcr;
  float* Cf; unsigned short* Cb; long long cB0, cB1; int ldc;
  int M, N, K;
  int zInner;
  int gelu;
};

#define BM 128
#define BN 64
#define LSTR 40   // LDS row stride (ushorts): 80B rows -> b128-aligned, conflict-spread

// fragment build + 4 WMMAs for one 32x32 wave sub-tile
__device__ __forceinline__ void frag_mma(const unsigned short (*lA)[LSTR],
                                         const unsigned short (*lB)[LSTR],
                                         int wm, int wn, int half, int l15,
                                         v8f acc[4]) {
  FragBF fa0, fa1, fb0, fb1;
  #pragma unroll
  for (int v = 0; v < 8; ++v) {
    const int kk = ((v < 4) ? (2 * v) : (16 + 2 * (v - 4))) + 8 * half;
    fa0.u[v] = *(const unsigned int*)&lA[wm      + l15][kk];
    fa1.u[v] = *(const unsigned int*)&lA[wm + 16 + l15][kk];
    fb0.u[v] = *(const unsigned int*)&lB[wn      + l15][kk];
    fb1.u[v] = *(const unsigned int*)&lB[wn + 16 + l15][kk];
  }
  acc[0] = __builtin_amdgcn_wmma_f32_16x16x32_bf16(false, fa0.v, false, fb0.v, (short)0, acc[0], false, false);
  acc[1] = __builtin_amdgcn_wmma_f32_16x16x32_bf16(false, fa0.v, false, fb1.v, (short)0, acc[1], false, false);
  acc[2] = __builtin_amdgcn_wmma_f32_16x16x32_bf16(false, fa1.v, false, fb0.v, (short)0, acc[2], false, false);
  acc[3] = __builtin_amdgcn_wmma_f32_16x16x32_bf16(false, fa1.v, false, fb1.v, (short)0, acc[3], false, false);
}

__global__ __launch_bounds__(256)
void k_gemm_wmma(GemmP p) {
  __shared__ __align__(16) unsigned short lA[BM][LSTR];
  __shared__ __align__(16) unsigned short lB[BN][LSTR];

  const int z  = blockIdx.z;
  const int zb = z / p.zInner;
  const int zi = z - zb * p.zInner;
  const unsigned short* A = p.A + zb * p.aB0 + zi * p.aB1;
  const unsigned short* B = p.B + zb * p.bB0 + zi * p.bB1;
  const float* bias = p.bias ? (p.bias + zb * p.biasB0 + zi * p.biasB1) : nullptr;
  const float* crs  = p.Cres ? (p.Cres + zb * p.crB0  + zi * p.crB1)  : nullptr;
  float*          cf = p.Cf ? (p.Cf + zb * p.cB0 + zi * p.cB1) : nullptr;
  unsigned short* cb = p.Cb ? (p.Cb + zb * p.cB0 + zi * p.cB1) : nullptr;

  const int m0 = blockIdx.y * BM;
  const int n0 = blockIdx.x * BN;
  const int t    = threadIdx.x;
  const int wave = t >> 5, lane = t & 31;
  const int wm = (wave & 3) * 32;     // 4 waves along M (32 rows each)
  const int wn = (wave >> 2) * 32;    // 2 waves along N (32 cols each)
  const int half = lane >> 4, l15 = lane & 15;

  v8f acc[4];
  #pragma unroll
  for (int i = 0; i < 4; ++i) acc[i] = (v8f){0.f,0.f,0.f,0.f,0.f,0.f,0.f,0.f};

  const int ar = t >> 2;          // 0..63 (A rows ar, ar+64; B row ar)
  const int ac = (t & 3) * 8;     // 8 consecutive k per thread
  const int br = t >> 3;          // 0..31: k-row for bTrans fast path
  const int bc = (t & 7) * 8;     // 8 consecutive n for bTrans fast path

  const bool fullM = (m0 + BM <= p.M);
  const bool fullN = (n0 + BN <= p.N);
  const bool vecOK = ((p.lda & 7) == 0) && ((p.ldb & 7) == 0);
  const int  Kf    = (fullM && fullN && vecOK) ? (p.K & ~31) : 0;

  int k0 = 0;
  // ---------- fast loop: fully interior, branch-free ----------
  for (; k0 < Kf; k0 += 32) {
    __builtin_prefetch((const void*)(A + (long long)(m0 + ar) * p.lda + (k0 + 32 + ac)), 0, 1);
    *(uint4*)&lA[ar][ac] =
      *(const uint4*)(A + (long long)(m0 + ar) * p.lda + (k0 + ac));
    *(uint4*)&lA[ar + 64][ac] =
      *(const uint4*)(A + (long long)(m0 + ar + 64) * p.lda + (k0 + ac));
    if (!p.bTrans) {
      *(uint4*)&lB[ar][ac] =
        *(const uint4*)(B + (long long)(n0 + ar) * p.ldb + (k0 + ac));
    } else {
      V16B v; v.q = *(const uint4*)(B + (long long)(k0 + br) * p.ldb + (n0 + bc));
      #pragma unroll
      for (int i = 0; i < 8; ++i) lB[bc + i][br] = v.s[i];
    }
    __syncthreads();
    frag_mma(lA, lB, wm, wn, half, l15, acc);
    __syncthreads();
  }
  // ---------- generic loop: clamped loads + select (no exec branching) ----------
  for (; k0 < p.K; k0 += 32) {
    {
      const int gm0 = m0 + ar,      gmc0 = imin(gm0, p.M - 1);
      const int gm1 = m0 + ar + 64, gmc1 = imin(gm1, p.M - 1);
      const int gn  = n0 + ar,      gnc  = imin(gn,  p.N - 1);
      #pragma unroll
      for (int i = 0; i < 8; ++i) {
        const int gk  = k0 + ac + i;
        const int gkc = imin(gk, p.K - 1);
        const bool kin = (gk < p.K);
        unsigned short va0 = A[(long long)gmc0 * p.lda + gkc];
        unsigned short va1 = A[(long long)gmc1 * p.lda + gkc];
        unsigned short vb  = p.bTrans ? B[(long long)gkc * p.ldb + gnc]
                                      : B[(long long)gnc * p.ldb + gkc];
        lA[ar][ac + i]      = (gm0 < p.M && kin) ? va0 : (unsigned short)0;
        lA[ar + 64][ac + i] = (gm1 < p.M && kin) ? va1 : (unsigned short)0;
        lB[ar][ac + i]      = (gn  < p.N && kin) ? vb  : (unsigned short)0;
      }
    }
    __syncthreads();
    frag_mma(lA, lB, wm, wn, half, l15, acc);
    __syncthreads();
  }

  // epilogue: D layout = VGPR r -> row r + 8*half, col = lane&15
  #pragma unroll
  for (int mi = 0; mi < 2; ++mi) {
    #pragma unroll
    for (int ni = 0; ni < 2; ++ni) {
      const v8f a = acc[mi * 2 + ni];
      #pragma unroll
      for (int r = 0; r < 8; ++r) {
        const int row = m0 + wm + mi * 16 + r + 8 * half;
        if (row >= p.M) continue;
        const int col = n0 + wn + ni * 16 + l15;
        if (col >= p.N) continue;
        float val = a[r];
        if (bias) val += bias[col];
        if (crs)  val += crs[(long long)row * p.ldcr + col];
        if (p.gelu) val = 0.5f * val * (1.0f + erff(val * 0.70710678118f));
        const long long o = (long long)row * p.ldc + col;
        if (cf) cf[o] = val;
        if (cb) cb[o] = f2bf(val);
      }
    }
  }
}

// ---------------- elementwise / reduction kernels ----------------
__global__ __launch_bounds__(256)
void k_f2bf(const float* __restrict__ src, unsigned short* __restrict__ dst, int n) {
  int i = blockIdx.x * 256 + threadIdx.x;
  if (i < n) dst[i] = f2bf(src[i]);
}

// [16,3,224,224] -> bf16 [16*196, 768] with (c,ph,pw) per-patch flatten order
__global__ __launch_bounds__(256)
void k_patchify(const float* __restrict__ img, unsigned short* __restrict__ out, int n) {
  int idx = blockIdx.x * 256 + threadIdx.x;
  if (idx >= n) return;
  int d  = idx % 768;
  int p  = (idx / 768) % 196;
  int b  = idx / (768 * 196);
  int c  = d / 256, rem = d % 256, ph = rem / 16, pw = rem % 16;
  int pi = p / 14, pj = p % 14;
  float v = img[(((long long)b * 3 + c) * 224 + (pi * 16 + ph)) * 224 + (pj * 16 + pw)];
  out[idx] = f2bf(v);
}

// row 0 per image := class_token + pos[0]; all rows += pos_embed
__global__ __launch_bounds__(256)
void k_clspos(float* __restrict__ x, const float* __restrict__ cls,
              const float* __restrict__ pos, int n) {
  int idx = blockIdx.x * 256 + threadIdx.x;
  if (idx >= n) return;
  int col = idx % 768;
  int s   = (idx / 768) % 197;
  if (s == 0) x[idx] = cls[col] + pos[col];
  else        x[idx] += pos[s * 768 + col];
}

__global__ __launch_bounds__(256)
void k_add(float* __restrict__ x, const float* __restrict__ y, int n) {
  int i = blockIdx.x * 256 + threadIdx.x;
  if (i < n) x[i] += y[i];
}

// one 256-thread block per 768-wide row; bf16 output
__global__ __launch_bounds__(256)
void k_layernorm(const float* __restrict__ x, const float* __restrict__ w,
                 const float* __restrict__ b, unsigned short* __restrict__ out, int D) {
  __shared__ float ws1[8], ws2[8];
  const int row = blockIdx.x;
  const float* xr = x + (long long)row * D;
  float s = 0.f, s2 = 0.f;
  for (int i = threadIdx.x; i < D; i += 256) { float v = xr[i]; s += v; s2 += v * v; }
  for (int m = 16; m >= 1; m >>= 1) { s += __shfl_xor(s, m, 32); s2 += __shfl_xor(s2, m, 32); }
  const int wave = threadIdx.x >> 5, lane = threadIdx.x & 31;
  if (lane == 0) { ws1[wave] = s; ws2[wave] = s2; }
  __syncthreads();
  if (wave == 0) {
    s  = (lane < 8) ? ws1[lane] : 0.f;
    s2 = (lane < 8) ? ws2[lane] : 0.f;
    for (int m = 4; m >= 1; m >>= 1) { s += __shfl_xor(s, m, 32); s2 += __shfl_xor(s2, m, 32); }
    if (lane == 0) { ws1[0] = s; ws2[0] = s2; }
  }
  __syncthreads();
  const float mu  = ws1[0] / D;
  const float var = ws2[0] / D - mu * mu;
  const float inv = rsqrtf(var + 1e-5f);
  unsigned short* orow = out + (long long)row * D;
  for (int i = threadIdx.x; i < D; i += 256)
    orow[i] = f2bf((xr[i] - mu) * inv * w[i] + b[i]);
}

// one wave per row: scaled softmax fp32(ldi) -> bf16(ldo)
__global__ __launch_bounds__(256)
void k_softmax(const float* __restrict__ S, int ldi,
               unsigned short* __restrict__ P, int ldo,
               int rows, int n, float scale) {
  const int row  = (blockIdx.x * 256 + threadIdx.x) >> 5;
  const int lane = threadIdx.x & 31;
  if (row >= rows) return;
  const float* sr = S + (long long)row * ldi;
  unsigned short* pr = P + (long long)row * ldo;
  float mx = -1e30f;
  for (int i = lane; i < n; i += 32) mx = fmaxf(mx, sr[i] * scale);
  for (int m = 16; m >= 1; m >>= 1) mx = fmaxf(mx, __shfl_xor(mx, m, 32));
  float sum = 0.f;
  for (int i = lane; i < n; i += 32) sum += expf(sr[i] * scale - mx);
  for (int m = 16; m >= 1; m >>= 1) sum += __shfl_xor(sum, m, 32);
  const float inv = 1.f / sum;
  for (int i = lane; i < n; i += 32)
    pr[i] = f2bf(expf(sr[i] * scale - mx) * inv);
}

// ---------------- host orchestration ----------------
static inline size_t alignUp(size_t x) { return (x + 255) & ~(size_t)255; }

extern "C" void kernel_launch(void* const* d_in, const int* in_sizes, int n_in,
                              void* d_out, int out_size, void* d_ws, size_t ws_size,
                              hipStream_t stream) {
  (void)in_sizes; (void)n_in; (void)out_size; (void)ws_size;
  const float* images = (const float*)d_in[0];
  const float* w_map  = (const float*)d_in[1];
  const float* b_map  = (const float*)d_in[2];
  const float* cls    = (const float*)d_in[3];
  const float* pos    = (const float*)d_in[4];
  const float* qw = (const float*)d_in[5];  const float* qb = (const float*)d_in[6];
  const float* kw = (const float*)d_in[7];  const float* kb = (const float*)d_in[8];
  const float* vw = (const float*)d_in[9];  const float* vb = (const float*)d_in[10];
  const float* ln1w = (const float*)d_in[11]; const float* ln1b = (const float*)d_in[12];
  const float* ln2w = (const float*)d_in[13]; const float* ln2b = (const float*)d_in[14];
  const float* w1 = (const float*)d_in[15]; const float* b1 = (const float*)d_in[16];
  const float* w2 = (const float*)d_in[17]; const float* b2 = (const float*)d_in[18];

  enum { BATCH = 16, SEQ = 197, HID = 768, DFF = 3072, NH = 12, DH = 64, LDP = 200 };
  const int TT   = BATCH * SEQ;          // 3152 tokens
  const int TOK  = TT * HID;             // 2420736
  const int PATN = BATCH * 196 * HID;    // 2408448
  const int QWN  = 12 * NH * DH * DH;    // 589824
  const int W1N  = DFF * HID;            // 2359296 (one block)
  const int SC1  = SEQ * SEQ;            // 38809
  const int SCN  = BATCH * NH * SC1;     // 7451328
  const int PROW = BATCH * NH * SEQ;     // 37824 probability rows (stride LDP)
  const int HMID = TT * DFF;             // 9682944

  char* wsb = (char*)d_ws; size_t off = 0;
  auto carve = [&](size_t bytes) -> void* { void* p = wsb + off; off = alignUp(off + bytes); return p; };
  unsigned short* wmap_bf = (unsigned short*)carve((size_t)QWN * 2);
  unsigned short* qw_bf   = (unsigned short*)carve((size_t)QWN * 2);
  unsigned short* kw_bf   = (unsigned short*)carve((size_t)QWN * 2);
  unsigned short* vw_bf   = (unsigned short*)carve((size_t)QWN * 2);
  unsigned short* w1_bf   = (unsigned short*)carve((size_t)W1N * 2);
  unsigned short* w2_bf   = (unsigned short*)carve((size_t)W1N * 2);
  unsigned short* Ppat    = (unsigned short*)carve((size_t)PATN * 2);
  float*          x       = (float*)carve((size_t)TOK * 4);
  unsigned short* hbf     = (unsigned short*)carve((size_t)TOK * 2);
  unsigned short* qf      = (unsigned short*)carve((size_t)TOK * 2);
  unsigned short* kf      = (unsigned short*)carve((size_t)TOK * 2);
  unsigned short* vf      = (unsigned short*)carve((size_t)TOK * 2);
  float*          sc      = (float*)carve((size_t)SCN * 4);
  unsigned short* Pbf     = (unsigned short*)carve((size_t)PROW * LDP * 2);
  float*          obuf    = (float*)carve((size_t)TOK * 4);
  unsigned short* hmid    = (unsigned short*)carve((size_t)HMID * 2);

  auto cvt = [&](const float* s, unsigned short* d, int n) {
    k_f2bf<<<dim3((n + 255) / 256), dim3(256), 0, stream>>>(s, d, n);
  };
  auto gemm = [&](const GemmP& g, int batch) {
    dim3 grid((g.N + BN - 1) / BN, (g.M + BM - 1) / BM, batch);
    k_gemm_wmma<<<grid, dim3(256), 0, stream>>>(g);
  };

  // weight conversion (small weights up front; MLP weights per block)
  cvt(w_map, wmap_bf, QWN);
  cvt(qw, qw_bf, QWN); cvt(kw, kw_bf, QWN); cvt(vw, vw_bf, QWN);

  // patchify + patch-embedding GEMM into x rows [1..196] per image
  k_patchify<<<dim3((PATN + 255) / 256), dim3(256), 0, stream>>>(images, Ppat, PATN);
  {
    GemmP g{}; g.A = Ppat; g.aB0 = 196 * HID; g.aB1 = 0; g.lda = HID;
    g.B = wmap_bf; g.bB0 = 0; g.bB1 = 0; g.ldb = HID; g.bTrans = 0;
    g.bias = b_map; g.biasB0 = 0; g.biasB1 = 0;
    g.Cres = nullptr; g.crB0 = g.crB1 = 0; g.ldcr = 0;
    g.Cf = x + HID; g.Cb = nullptr; g.cB0 = (long long)SEQ * HID; g.cB1 = 0; g.ldc = HID;
    g.M = 196; g.N = HID; g.K = HID; g.zInner = 1; g.gelu = 0;
    gemm(g, BATCH);
  }
  k_clspos<<<dim3((TOK + 255) / 256), dim3(256), 0, stream>>>(x, cls, pos, TOK);

  for (int l = 0; l < 12; ++l) {
    // LN1
    k_layernorm<<<dim3(TT), dim3(256), 0, stream>>>(x, ln1w + l * HID, ln1b + l * HID, hbf, HID);

    // per-head QKV projections (batched over 12 heads)
    const unsigned short* wbf[3] = { qw_bf, kw_bf, vw_bf };
    const float* bv[3] = { qb, kb, vb };
    unsigned short* outv[3] = { qf, kf, vf };
    for (int m = 0; m < 3; ++m) {
      GemmP g{}; g.A = hbf; g.aB0 = 0; g.aB1 = DH; g.lda = HID;
      g.B = wbf[m] + (long long)l * NH * DH * DH; g.bB0 = 0; g.bB1 = DH * DH; g.ldb = DH; g.bTrans = 0;
      g.bias = bv[m] + (long long)l * NH * DH; g.biasB0 = 0; g.biasB1 = DH;
      g.Cres = nullptr; g.crB0 = g.crB1 = 0; g.ldcr = 0;
      g.Cf = nullptr; g.Cb = outv[m]; g.cB0 = 0; g.cB1 = DH; g.ldc = HID;
      g.M = TT; g.N = DH; g.K = DH; g.zInner = NH; g.gelu = 0;
      gemm(g, NH);
    }

    // scores = Q @ K^T per (b,h)   [197x197, K=64]
    {
      GemmP g{}; g.A = qf; g.aB0 = (long long)SEQ * HID; g.aB1 = DH; g.lda = HID;
      g.B = kf; g.bB0 = (long long)SEQ * HID; g.bB1 = DH; g.ldb = HID; g.bTrans = 0;
      g.bias = nullptr; g.biasB0 = g.biasB1 = 0;
      g.Cres = nullptr; g.crB0 = g.crB1 = 0; g.ldcr = 0;
      g.Cf = sc; g.Cb = nullptr; g.cB0 = (long long)NH * SC1; g.cB1 = SC1; g.ldc = SEQ;
      g.M = SEQ; g.N = SEQ; g.K = DH; g.zInner = NH; g.gelu = 0;
      gemm(g, BATCH * NH);
    }
    // softmax (scale 1/8 folded in), P -> bf16 with padded stride LDP
    k_softmax<<<dim3((PROW * 32 + 255) / 256), dim3(256), 0, stream>>>(
        sc, SEQ, Pbf, LDP, PROW, SEQ, 0.125f);

    // O = P @ V per (b,h)  (Btrans: B[k,n] = V[t,d])
    {
      GemmP g{}; g.A = Pbf; g.aB0 = (long long)NH * SEQ * LDP; g.aB1 = (long long)SEQ * LDP; g.lda = LDP;
      g.B = vf; g.bB0 = (long long)SEQ * HID; g.bB1 = DH; g.ldb = HID; g.bTrans = 1;
      g.bias = nullptr; g.biasB0 = g.biasB1 = 0;
      g.Cres = nullptr; g.crB0 = g.crB1 = 0; g.ldcr = 0;
      g.Cf = obuf; g.Cb = nullptr; g.cB0 = (long long)SEQ * HID; g.cB1 = DH; g.ldc = HID;
      g.M = SEQ; g.N = DH; g.K = SEQ; g.zInner = NH; g.gelu = 0;
      gemm(g, BATCH * NH);
    }
    // residual
    k_add<<<dim3((TOK + 255) / 256), dim3(256), 0, stream>>>(x, obuf, TOK);

    // LN2
    k_layernorm<<<dim3(TT), dim3(256), 0, stream>>>(x, ln2w + l * HID, ln2b + l * HID, hbf, HID);

    // MLP weights for this block -> bf16 (fits/streams from L2)
    cvt(w1 + (long long)l * W1N, w1_bf, W1N);
    cvt(w2 + (long long)l * W1N, w2_bf, W1N);

    // MLP1 + exact GELU -> bf16 hidden
    {
      GemmP g{}; g.A = hbf; g.aB0 = g.aB1 = 0; g.lda = HID;
      g.B = w1_bf; g.bB0 = g.bB1 = 0; g.ldb = HID; g.bTrans = 0;
      g.bias = b1 + (long long)l * DFF; g.biasB0 = g.biasB1 = 0;
      g.Cres = nullptr; g.crB0 = g.crB1 = 0; g.ldcr = 0;
      g.Cf = nullptr; g.Cb = hmid; g.cB0 = g.cB1 = 0; g.ldc = DFF;
      g.M = TT; g.N = DFF; g.K = HID; g.zInner = 1; g.gelu = 1;
      gemm(g, 1);
    }
    // MLP2 + residual (last block writes d_out)
    {
      GemmP g{}; g.A = hmid; g.aB0 = g.aB1 = 0; g.lda = DFF;
      g.B = w2_bf; g.bB0 = g.bB1 = 0; g.ldb = DFF; g.bTrans = 0;
      g.bias = b2 + (long long)l * HID; g.biasB0 = g.biasB1 = 0;
      g.Cres = x; g.crB0 = g.crB1 = 0; g.ldcr = HID;
      g.Cf = (l == 11) ? (float*)d_out : x; g.Cb = nullptr; g.cB0 = g.cB1 = 0; g.ldc = HID;
      g.M = TT; g.N = HID; g.K = DFF; g.zInner = 1; g.gelu = 0;
      gemm(g, 1);
    }
  }
}
